// RbMLayer_19825569038535
// MI455X (gfx1250) — matrix operand order repaired and checked by reading, not compiled, not run on hardware
//
#include <hip/hip_runtime.h>
#include <cstdint>
#include <cstddef>

#define E_    8
#define K_    3
#define D_    1024
#define DFF_  4096
#define N_    8192
#define MT_   32     // tokens per FFN block (two 16-row WMMA tiles)
#define XSTR  1032   // 1024 + 8 halves pad (keeps 16B alignment, breaks bank conflicts)
#define HSTR  72     // 64 + 8 halves pad
#define CLAMP_MAX_ 4.605170185988091f

typedef __bf16 bf16;
typedef __attribute__((ext_vector_type(16))) bf16  v16bf;
typedef __attribute__((ext_vector_type(8)))  float v8f;

union ABfrag { v16bf v; uint4 u[2]; };

#if __has_builtin(__builtin_amdgcn_global_load_async_to_lds_b128) && \
    __has_builtin(__builtin_amdgcn_s_wait_asynccnt)
#define HAVE_ASYNC 1
typedef int v4i_async __attribute__((vector_size(16)));
typedef __attribute__((address_space(1))) v4i_async* g4_ptr;
typedef __attribute__((address_space(3))) v4i_async* l4_ptr;
#else
#define HAVE_ASYNC 0
#endif

// ---------------- 1. x -> bf16 ----------------
__global__ void cvt_x_kernel(const float* __restrict__ x, bf16* __restrict__ xb) {
  size_t i = (size_t)blockIdx.x * 256 + threadIdx.x;
  if (i < (size_t)N_ * D_) xb[i] = (bf16)x[i];
}

// ---------------- 2. transpose + convert weights ----------------
// in: [z][rows][cols] f32  ->  out: [z][cols][rows] bf16
__global__ void transpose_cvt_kernel(const float* __restrict__ in, bf16* __restrict__ out,
                                     int rows, int cols) {
  __shared__ float tile[32][33];
  int z = blockIdx.z;
  const float* ib = in + (size_t)z * rows * cols;
  bf16* ob = out + (size_t)z * rows * cols;
  int colBase = blockIdx.x * 32, rowBase = blockIdx.y * 32;
  int tx = threadIdx.x, ty = threadIdx.y;
#pragma unroll
  for (int i = 0; i < 32; i += 8)
    tile[ty + i][tx] = ib[(size_t)(rowBase + ty + i) * cols + colBase + tx];
  __syncthreads();
#pragma unroll
  for (int i = 0; i < 32; i += 8)
    ob[(size_t)(colBase + ty + i) * rows + rowBase + tx] = (bf16)tile[tx][ty + i];
}

// ---------------- 3. normalize routing memory + SIM aux ----------------
__global__ void rmnorm_sim_kernel(const float* __restrict__ rm, float* __restrict__ rmn,
                                  float* __restrict__ auxbuf) {
  __shared__ float simsh[64];
  int wave = threadIdx.x >> 5, lane = threadIdx.x & 31;
  const float* r = rm + (size_t)wave * D_;
  float ss = 0.f;
  for (int j = lane; j < D_; j += 32) { float v = r[j]; ss += v * v; }
#pragma unroll
  for (int off = 16; off > 0; off >>= 1) ss += __shfl_xor(ss, off, 32);
  float inv = 1.f / fmaxf(sqrtf(ss), 1e-12f);
  for (int j = lane; j < D_; j += 32) rmn[(size_t)wave * D_ + j] = r[j] * inv;
  __syncthreads();
#pragma unroll
  for (int jr = 0; jr < E_; jr++) {
    float d = 0.f;
    for (int j = lane; j < D_; j += 32)
      d += rmn[(size_t)wave * D_ + j] * rmn[(size_t)jr * D_ + j];
#pragma unroll
    for (int off = 16; off > 0; off >>= 1) d += __shfl_xor(d, off, 32);
    if (lane == 0) simsh[wave * 8 + jr] = d;
  }
  __syncthreads();
  if (threadIdx.x == 0) {
    float s = 0.f;
    for (int i = 0; i < 64; i++) s += simsh[i];
    auxbuf[0] = 0.01f * (s / 64.f);   // SIM * mean(rm_n @ rm_n.T)
  }
}

// ---------------- 4. routing: dists, top-3, gates, lists ----------------
__global__ void route_kernel(const float* __restrict__ x, const float* __restrict__ rmn,
                             float* __restrict__ gates, float* __restrict__ att,
                             float* __restrict__ vq, int* __restrict__ cnt,
                             int* __restrict__ lists) {
  __shared__ float rs[E_ * D_];
  for (int i = threadIdx.x; i < E_ * D_; i += 256) rs[i] = rmn[i];
  __syncthreads();
  int wave = threadIdx.x >> 5, lane = threadIdx.x & 31;
  int n = blockIdx.x * 8 + wave;
  const float* xr = x + (size_t)n * D_;
  float dot[E_];
#pragma unroll
  for (int e = 0; e < E_; e++) dot[e] = 0.f;
  float ss = 0.f;
  for (int j = lane; j < D_; j += 32) {
    float xv = xr[j]; ss += xv * xv;
#pragma unroll
    for (int e = 0; e < E_; e++) dot[e] += xv * rs[e * D_ + j];
  }
#pragma unroll
  for (int off = 16; off > 0; off >>= 1) {
    ss += __shfl_xor(ss, off, 32);
#pragma unroll
    for (int e = 0; e < E_; e++) dot[e] += __shfl_xor(dot[e], off, 32);
  }
  if (lane == 0) {
    float inv = 1.f / fmaxf(sqrtf(ss), 1e-12f);
    float dist[E_];
#pragma unroll
    for (int e = 0; e < E_; e++) dist[e] = dot[e] * inv;
    int idx[K_]; float val[K_]; bool used[E_];
#pragma unroll
    for (int e = 0; e < E_; e++) used[e] = false;
#pragma unroll
    for (int k = 0; k < K_; k++) {
      float best = -3.4e38f; int bi = 0;
#pragma unroll
      for (int e = 0; e < E_; e++)
        if (!used[e] && dist[e] > best) { best = dist[e]; bi = e; }
      used[bi] = true; idx[k] = bi; val[k] = best;
    }
    float m = val[0];
    float e0 = expf(val[0] - m), e1 = expf(val[1] - m), e2 = expf(val[2] - m);
    float sinv = 1.f / (e0 + e1 + e2);
    float g[K_] = {e0 * sinv, e1 * sinv, e2 * sinv};
    float a8[E_];
#pragma unroll
    for (int e = 0; e < E_; e++) a8[e] = 0.f;
#pragma unroll
    for (int k = 0; k < K_; k++) a8[idx[k]] = g[k];
#pragma unroll
    for (int e = 0; e < E_; e++) att[(size_t)n * E_ + e] = a8[e];
    float vqs = 0.f;
#pragma unroll
    for (int k = 0; k < K_; k++) {
      gates[n * K_ + k] = g[k];
      vqs += g[k] * val[k];
      int pos = atomicAdd(&cnt[idx[k]], 1);
      lists[idx[k] * N_ + pos] = (n << 2) | k;
    }
    vq[n] = -vqs;
  }
}

// ---------------- 5. aux reduction (VQ + SIM + load-balance) ----------------
__global__ void aux_kernel(const float* __restrict__ att, const float* __restrict__ vq,
                           const float* __restrict__ auxbuf, float* __restrict__ out_aux) {
  __shared__ float sh[256];
  __shared__ float loadsh[E_];
  int tid = threadIdx.x, wave = tid >> 5, lane = tid & 31;
  float p = 0.f;
  for (int n = lane; n < N_; n += 32) p += att[(size_t)n * E_ + wave];
#pragma unroll
  for (int off = 16; off > 0; off >>= 1) p += __shfl_xor(p, off, 32);
  if (lane == 0) loadsh[wave] = p;
  float vp = 0.f;
  for (int i = tid; i < N_; i += 256) vp += vq[i];
  sh[tid] = vp;
  __syncthreads();
  for (int s = 128; s > 0; s >>= 1) {
    if (tid < s) sh[tid] += sh[tid + s];
    __syncthreads();
  }
  if (tid == 0) {
    float mean = 0.f;
    for (int e = 0; e < E_; e++) mean += loadsh[e];
    mean *= (1.f / E_);
    float var = 0.f;
    for (int e = 0; e < E_; e++) { float d = loadsh[e] - mean; var += d * d; }
    var *= (1.f / (E_ - 1));                       // ddof=1
    float cv2 = var / (mean * mean + 1e-10f);
    *out_aux = 0.05f * (sh[0] / N_) + auxbuf[0] + 0.01f * cv2;
  }
}

// ---------------- 6. grouped expert FFN (bf16 WMMA, 32-token M tile) ----------------
__global__ __launch_bounds__(128) void ffn_kernel(
    const bf16* __restrict__ xb, const bf16* __restrict__ w1t, const bf16* __restrict__ w2t,
    const float* __restrict__ b1, const float* __restrict__ b2,
    const float* __restrict__ gates, const int* __restrict__ cnt,
    const int* __restrict__ lists, float* __restrict__ comb) {
  int e = blockIdx.x >> 8;          // 256 tiles per expert, expert-major for L2 reuse
  int tile = blockIdx.x & 255;
  int count = cnt[e];
  if (tile * MT_ >= count) return;  // block-uniform exit

  __shared__ bf16 xs[MT_ * XSTR];
  __shared__ bf16 hs[MT_ * HSTR];
  __shared__ int toks[MT_];
  __shared__ int slotoff[MT_];
  __shared__ float grow[MT_];

  int tid = threadIdx.x;
  if (tid < MT_) {
    int g = tile * MT_ + tid;
    if (g < count) {
      int pk = lists[e * N_ + g];
      int tok = pk >> 2, sl = pk & 3;
      toks[tid] = tok;
      slotoff[tid] = tok * K_ + sl;
      grow[tid] = gates[tok * K_ + sl];
    } else {
      toks[tid] = 0; slotoff[tid] = -1; grow[tid] = 0.f;
    }
  }
  __syncthreads();
  { // stage 32 gathered token rows (bf16) into LDS: 4 threads/row, 512B per thread
    int r = tid >> 2, p = tid & 3;
    const bf16* srow = xb + (size_t)toks[r] * D_ + p * 256;
    bf16* drow = &xs[(size_t)r * XSTR + p * 256];
#if HAVE_ASYNC
    g4_ptr gs = (g4_ptr)srow;
    l4_ptr ld = (l4_ptr)drow;
#pragma unroll
    for (int j = 0; j < 32; j++)
      __builtin_amdgcn_global_load_async_to_lds_b128(gs + j, ld + j, 0, 0);
    __builtin_amdgcn_s_wait_asynccnt(0);
#else
    const uint4* src = reinterpret_cast<const uint4*>(srow);
    uint4* dst = reinterpret_cast<uint4*>(drow);
#pragma unroll
    for (int j = 0; j < 32; j++) dst[j] = src[j];
#endif
  }
  __syncthreads();

  int wave = tid >> 5, lane = tid & 31;
  int laneq = lane & 15;
  bool hi = lane >= 16;

  v8f accA[16], accB[16];
#pragma unroll
  for (int t = 0; t < 16; t++) { accA[t] = v8f{}; accB[t] = v8f{}; }

  const bf16* ap1 = &xs[(size_t)laneq * XSTR + (hi ? 8 : 0)];
  const bf16* ap2 = &hs[(size_t)laneq * HSTR + (hi ? 8 : 0)];

#pragma unroll 1
  for (int c = 0; c < DFF_ / 64; c++) {
    // ---- GEMM1: h[32 x 16 per wave] = x(32x1024) @ W1[:, chunk cols] ----
    int n1 = c * 64 + wave * 16 + laneq;
    const bf16* bp1 = w1t + ((size_t)e * DFF_ + n1) * D_ + (hi ? 8 : 0);
    __builtin_prefetch(bp1 + (size_t)D_ * 64, 0, 1);   // next chunk's W1 column
    v8f h0 = v8f{}, h1 = v8f{};
#pragma unroll 4
    for (int kk = 0; kk < D_; kk += 32) {
      ABfrag b, aA, aB;
      b.u[0] = *reinterpret_cast<const uint4*>(bp1 + kk);        // K 8h..8h+7, 16+8h..16+8h+7
      b.u[1] = *reinterpret_cast<const uint4*>(bp1 + kk + 16);
      aA.u[0] = *reinterpret_cast<const uint4*>(ap1 + kk);       // rows 0..15
      aA.u[1] = *reinterpret_cast<const uint4*>(ap1 + kk + 16);
      aB.u[0] = *reinterpret_cast<const uint4*>(ap1 + 16 * XSTR + kk);   // rows 16..31
      aB.u[1] = *reinterpret_cast<const uint4*>(ap1 + 16 * XSTR + kk + 16);
      h0 = __builtin_amdgcn_wmma_f32_16x16x32_bf16(false, aA.v, false, b.v,
                                                   (short)0, h0, false, false);
      h1 = __builtin_amdgcn_wmma_f32_16x16x32_bf16(false, aB.v, false, b.v,
                                                   (short)0, h1, false, false);
    }
    float b1v = b1[e * DFF_ + n1];
#pragma unroll
    for (int v = 0; v < 8; v++) {
      int m = v + (hi ? 8 : 0);
      hs[m * HSTR + wave * 16 + laneq] = (bf16)fmaxf(h0[v] + b1v, 0.f);
      hs[(m + 16) * HSTR + wave * 16 + laneq] = (bf16)fmaxf(h1[v] + b1v, 0.f);
    }
    __syncthreads();
    // ---- GEMM2: o[32 x 256 per wave] += h(32x64) @ W2[chunk rows, :] ----
    ABfrag a0A, a1A, a0B, a1B;
    a0A.u[0] = *reinterpret_cast<const uint4*>(ap2 + 0);
    a0A.u[1] = *reinterpret_cast<const uint4*>(ap2 + 16);
    a1A.u[0] = *reinterpret_cast<const uint4*>(ap2 + 32);
    a1A.u[1] = *reinterpret_cast<const uint4*>(ap2 + 48);
    a0B.u[0] = *reinterpret_cast<const uint4*>(ap2 + 16 * HSTR + 0);
    a0B.u[1] = *reinterpret_cast<const uint4*>(ap2 + 16 * HSTR + 16);
    a1B.u[0] = *reinterpret_cast<const uint4*>(ap2 + 16 * HSTR + 32);
    a1B.u[1] = *reinterpret_cast<const uint4*>(ap2 + 16 * HSTR + 48);
#pragma unroll
    for (int t = 0; t < 16; t++) {
      int n2 = wave * 256 + t * 16 + laneq;
      const bf16* bp2 = w2t + ((size_t)e * D_ + n2) * DFF_ + c * 64 + (hi ? 8 : 0);
      ABfrag b0, b1f;
      b0.u[0] = *reinterpret_cast<const uint4*>(bp2);
      b0.u[1] = *reinterpret_cast<const uint4*>(bp2 + 16);
      b1f.u[0] = *reinterpret_cast<const uint4*>(bp2 + 32);
      b1f.u[1] = *reinterpret_cast<const uint4*>(bp2 + 48);
      accA[t] = __builtin_amdgcn_wmma_f32_16x16x32_bf16(false, a0A.v, false, b0.v,
                                                        (short)0, accA[t], false, false);
      accA[t] = __builtin_amdgcn_wmma_f32_16x16x32_bf16(false, a1A.v, false, b1f.v,
                                                        (short)0, accA[t], false, false);
      accB[t] = __builtin_amdgcn_wmma_f32_16x16x32_bf16(false, a0B.v, false, b0.v,
                                                        (short)0, accB[t], false, false);
      accB[t] = __builtin_amdgcn_wmma_f32_16x16x32_bf16(false, a1B.v, false, b1f.v,
                                                        (short)0, accB[t], false, false);
    }
    __syncthreads();
  }

  // epilogue: comb[(tok*3+slot)*D + col] = gate * (o + b2)
#pragma unroll
  for (int t = 0; t < 16; t++) {
    int col = wave * 256 + t * 16 + laneq;
    float b2v = b2[e * D_ + col];
#pragma unroll
    for (int v = 0; v < 8; v++) {
      int m = v + (hi ? 8 : 0);
      int so = slotoff[m];
      if (so >= 0)
        comb[(size_t)so * D_ + col] = grow[m] * (accA[t][v] + b2v);
      int so2 = slotoff[m + 16];
      if (so2 >= 0)
        comb[(size_t)so2 * D_ + col] = grow[m + 16] * (accB[t][v] + b2v);
    }
  }
}

// ---------------- 7. combine slots + logit scale ----------------
__global__ void combine_kernel(const float* __restrict__ comb, const float* __restrict__ temp,
                               float* __restrict__ out) {
  size_t i = (size_t)blockIdx.x * 256 + threadIdx.x;
  size_t nvec = (size_t)N_ * D_ / 4;
  if (i >= nvec) return;
  float scale = expf(fminf(temp[0], CLAMP_MAX_));
  size_t n = i / (D_ / 4), dv = i % (D_ / 4);
  const float4* c = reinterpret_cast<const float4*>(comb);
  float4 a = c[(n * 3 + 0) * (D_ / 4) + dv];
  float4 b = c[(n * 3 + 1) * (D_ / 4) + dv];
  float4 d = c[(n * 3 + 2) * (D_ / 4) + dv];
  float4 r;
  r.x = scale * (a.x + b.x + d.x);
  r.y = scale * (a.y + b.y + d.y);
  r.z = scale * (a.z + b.z + d.z);
  r.w = scale * (a.w + b.w + d.w);
  reinterpret_cast<float4*>(out)[i] = r;
}

extern "C" void kernel_launch(void* const* d_in, const int* in_sizes, int n_in,
                              void* d_out, int out_size, void* d_ws, size_t ws_size,
                              hipStream_t stream) {
  const float* x   = (const float*)d_in[0];
  const float* rm  = (const float*)d_in[1];
  const float* tmp = (const float*)d_in[2];
  const float* W1  = (const float*)d_in[3];
  const float* b1  = (const float*)d_in[4];
  const float* W2  = (const float*)d_in[5];
  const float* b2  = (const float*)d_in[6];
  float* out = (float*)d_out;
  char* ws = (char*)d_ws;
  (void)in_sizes; (void)n_in; (void)out_size; (void)ws_size;

  size_t off = 0;
  auto take = [&](size_t b) { size_t o = off; off += (b + 255) & ~(size_t)255; return o; };
  float* rmn    = (float*)(ws + take((size_t)E_ * D_ * 4));
  float* att    = (float*)(ws + take((size_t)N_ * E_ * 4));
  float* gates  = (float*)(ws + take((size_t)N_ * K_ * 4));
  float* vq     = (float*)(ws + take((size_t)N_ * 4));
  float* auxbuf = (float*)(ws + take(256));
  int*   cnt    = (int*)  (ws + take((size_t)E_ * 4));
  int*   lists  = (int*)  (ws + take((size_t)E_ * N_ * 4));
  bf16*  xb     = (bf16*) (ws + take((size_t)N_ * D_ * 2));
  bf16*  w1t    = (bf16*) (ws + take((size_t)E_ * D_ * DFF_ * 2));
  bf16*  w2t    = (bf16*) (ws + take((size_t)E_ * D_ * DFF_ * 2));
  float* comb   = (float*)(ws + take((size_t)N_ * K_ * D_ * 4));

  cvt_x_kernel<<<(N_ * D_ + 255) / 256, 256, 0, stream>>>(x, xb);
  dim3 tb(32, 8);
  transpose_cvt_kernel<<<dim3(DFF_ / 32, D_ / 32, E_), tb, 0, stream>>>(W1, w1t, D_, DFF_);
  transpose_cvt_kernel<<<dim3(D_ / 32, DFF_ / 32, E_), tb, 0, stream>>>(W2, w2t, DFF_, D_);
  rmnorm_sim_kernel<<<1, 256, 0, stream>>>(rm, rmn, auxbuf);
  (void)hipMemsetAsync(cnt, 0, E_ * sizeof(int), stream);
  route_kernel<<<N_ / 8, 256, 0, stream>>>(x, rmn, gates, att, vq, cnt, lists);
  aux_kernel<<<1, 256, 0, stream>>>(att, vq, auxbuf, out + (size_t)N_ * D_);
  ffn_kernel<<<E_ * (N_ / MT_), 128, 0, stream>>>(xb, w1t, w2t, b1, b2, gates, cnt, lists, comb);
  combine_kernel<<<((N_ * D_ / 4) + 255) / 256, 256, 0, stream>>>(comb, tmp, out);
}
